// RoIAlign3d_3762391351871
// MI455X (gfx1250) — matrix operand order repaired and checked by reading, not compiled, not run on hardware
//
#include <hip/hip_runtime.h>

// RoIAlign-3D (crop_and_resize_3d, transform_fpcoor=True) for MI455X gfx1250.
// Strategy: per (box, 16-channel tile) wave. The d-axis trilinear interpolation
// is expressed as a 16x16x32 f16 WMMA: A = (h,w-weight scaled) 7x32 d-weight
// matrix, B = full 32-deep d-rows of 16 channels (one 128B line per channel),
// accumulated in f32. 4 corners per output (i,j) -> 4 chained WMMAs.
// All 16 row loads (4 corners x 4 float4) are issued before any consumption to
// maximize per-wave memory-level parallelism.

typedef __attribute__((ext_vector_type(16))) _Float16 v16h;
typedef __attribute__((ext_vector_type(8)))  float    v8f;

#define CROP 7
#define NB   2
#define CC   256
#define HH   64
#define WW   64
#define DD   32
#define MB   256
#define CT   16   // channels per wave tile

__global__ __launch_bounds__(256)
void roialign3d_wmma_kernel(const float* __restrict__ fm,
                            const float* __restrict__ boxes,
                            const int*   __restrict__ box_ind,
                            float*       __restrict__ out)
{
    const int lane = threadIdx.x & 31;
    const int gw   = blockIdx.x * 8 + (threadIdx.x >> 5);  // global wave id, 0..4095
    const int m    = gw >> 4;                              // box index 0..255
    const int ct   = gw & 15;                              // channel tile 0..15
    const int cbase = ct * CT;

    // ---- box parameters (uniform within wave) ----
    const float x1 = boxes[m * 6 + 0], y1 = boxes[m * 6 + 1], z1 = boxes[m * 6 + 2];
    const float x2 = boxes[m * 6 + 3], y2 = boxes[m * 6 + 4], z2 = boxes[m * 6 + 5];
    const int   n  = box_ind[m];
    const float sw = (x2 - x1) * (1.0f / CROP);
    const float sh = (y2 - y1) * (1.0f / CROP);
    const float sd = (z2 - z1) * (1.0f / CROP);

    // ---- per-lane d-axis row parameters: this lane owns output row k = lane&15 ----
    const int krow = lane & 15;
    const float dc  = z1 + ((float)krow + 0.5f) * sd - 0.5f;   // source d coord
    const float dfl = floorf(dc);
    const float fd  = dc - dfl;
    const int d0 = (int)fminf(fmaxf(dfl,        0.0f), (float)(DD - 1));
    const int d1 = (int)fminf(fmaxf(dfl + 1.0f, 0.0f), (float)(DD - 1));
    const float vd  = (dc >= 0.0f && dc <= (float)(DD - 1)) ? 1.0f : 0.0f;
    const float wd0 = (1.0f - fd) * vd;
    const float wd1 = fd * vd;
    const bool rowActive = (krow < CROP);

    // ---- base A matrix (f16 16x32): A[k, dsrc] = d-axis lerp weights ----
    // Layout (16-bit A 16x32, contiguous to match B): lane<16 -> K = {0..15},
    // lane>=16 -> K = {16..31}; element e of v16h = VGPR e/2 half e%2.
    const int kofs = (lane < 16) ? 0 : 16;
    v16h abase;
#pragma unroll
    for (int e = 0; e < 16; ++e) {
        const int K = e + kofs;
        float a = 0.0f;
        if (rowActive) {
            if (K == d0) a += wd0;
            if (K == d1) a += wd1;   // d0==d1 at clipped edges -> weights sum
        }
        abase[e] = (_Float16)a;
    }

    // ---- B-matrix source pointer: lane -> channel column + d-half ----
    // Layout (B 32x16 f16): lanes 0..15 = cols, K=0..15; lanes 16..31 = cols, K=16..31.
    const int ccol = lane & 15;
    const long long chanBase =
        ((long long)n * CC + (cbase + ccol)) * (long long)(HH * WW * DD);
    const float* fptr = fm + chanBase + ((lane < 16) ? 0 : 16);

    float* outc = out + (size_t)(m * CC + cbase + ccol) * (size_t)(CROP * CROP * CROP);

    for (int i = 0; i < CROP; ++i) {
        const float hcf = y1 + ((float)i + 0.5f) * sh - 0.5f;
        const float hfl = floorf(hcf);
        const float fh  = hcf - hfl;
        const int h0 = (int)fminf(fmaxf(hfl,        0.0f), (float)(HH - 1));
        const int h1 = (int)fminf(fmaxf(hfl + 1.0f, 0.0f), (float)(HH - 1));
        const float vh  = (hcf >= 0.0f && hcf <= (float)(HH - 1)) ? 1.0f : 0.0f;
        const float wh0 = (1.0f - fh) * vh;
        const float wh1 = fh * vh;

        for (int j = 0; j < CROP; ++j) {
            const float wcf = x1 + ((float)j + 0.5f) * sw - 0.5f;
            const float wfl = floorf(wcf);
            const float fw  = wcf - wfl;
            const int w0 = (int)fminf(fmaxf(wfl,        0.0f), (float)(WW - 1));
            const int w1 = (int)fminf(fmaxf(wfl + 1.0f, 0.0f), (float)(WW - 1));
            const float vw  = (wcf >= 0.0f && wcf <= (float)(WW - 1)) ? 1.0f : 0.0f;
            const float ww0 = (1.0f - fw) * vw;
            const float ww1 = fw * vw;

            const int   hIdx[4] = {h0, h0, h1, h1};
            const int   wIdx[4] = {w0, w1, w0, w1};
            const float cwt[4]  = {wh0 * ww0, wh0 * ww1, wh1 * ww0, wh1 * ww1};

            // ---- Phase 1: issue ALL 16 row loads (4 corners x 64B/lane) ----
            float4 raw[4][4];
#pragma unroll
            for (int corner = 0; corner < 4; ++corner) {
                const float4* p = (const float4*)(fptr +
                    (size_t)(hIdx[corner] * WW + wIdx[corner]) * DD);
                raw[corner][0] = p[0];
                raw[corner][1] = p[1];
                raw[corner][2] = p[2];
                raw[corner][3] = p[3];
            }

            // ---- Phase 2: convert -> scale A -> WMMA, per corner ----
            v8f acc = {0.f, 0.f, 0.f, 0.f, 0.f, 0.f, 0.f, 0.f};
#pragma unroll
            for (int corner = 0; corner < 4; ++corner) {
                v16h b;
#pragma unroll
                for (int q = 0; q < 4; ++q) {
                    b[4 * q + 0] = (_Float16)raw[corner][q].x;
                    b[4 * q + 1] = (_Float16)raw[corner][q].y;
                    b[4 * q + 2] = (_Float16)raw[corner][q].z;
                    b[4 * q + 3] = (_Float16)raw[corner][q].w;
                }
                const _Float16 hs = (_Float16)cwt[corner];
                v16h hsplat;
#pragma unroll
                for (int e = 0; e < 16; ++e) hsplat[e] = hs;
                v16h as = abase * hsplat;   // 8x v_pk_mul_f16
                // D(16x16 f32) += A(16x32 f16) x B(32x16 f16)
                acc = __builtin_amdgcn_wmma_f32_16x16x32_f16(
                    false, as, false, b, (short)0, acc, false, false);
            }

            // D layout: lanes 0..15 hold N=lane (channel), VGPR r = row M=r = output k.
            if (lane < 16) {
                float* op = outc + (size_t)(i * CROP + j) * CROP;
#pragma unroll
                for (int k = 0; k < CROP; ++k) op[k] = acc[k];
            }
        }
    }
}

extern "C" void kernel_launch(void* const* d_in, const int* in_sizes, int n_in,
                              void* d_out, int out_size, void* d_ws, size_t ws_size,
                              hipStream_t stream) {
    const float* fm      = (const float*)d_in[0];   // [2,256,64,64,32] f32
    const float* boxes   = (const float*)d_in[1];   // [256,6] f32
    const int*   box_ind = (const int*)  d_in[2];   // [256] i32
    float*       out     = (float*)d_out;           // [256,256,7,7,7] f32

    // 4096 waves = 256 boxes x 16 channel-tiles; 8 waves per 256-thread block.
    dim3 grid(512), block(256);
    roialign3d_wmma_kernel<<<grid, block, 0, stream>>>(fm, boxes, box_ind, out);
}
